// JointNetwork_82480551953124
// MI455X (gfx1250) — matrix operand order repaired and checked by reading, not compiled, not run on hardware
//
#include <hip/hip_runtime.h>

typedef _Float16 h16;
typedef __attribute__((ext_vector_type(16))) _Float16 v16h;
typedef __attribute__((ext_vector_type(8)))  _Float16 v8h;
typedef __attribute__((ext_vector_type(8)))  float    v8f;

#define LDA 648   // 640 + 8 halfs pad: row stride 1296B = 324 dwords -> 4-bank rotation per row

union AFrag { v16h v; struct { v8h lo, hi; } p; };

// A fragment (16x32 f16) from LDS, row-major tile with stride LDA halfs.
// K order used here (contiguous): lanes 0-15 -> M=lane, halfs 0..15 = K 0..15
//             lanes 16-31 -> M=lane-16, halfs 0..15 = K 16..31
__device__ __forceinline__ v16h load_a_lds(const h16* sh, int k0, int lane) {
  const int m    = lane & 15;
  const int koff = (lane >> 4) << 4;          // 0 or 16
  const h16* p = sh + m * LDA + k0 + koff;
  AFrag f;
  f.p.lo = *(const v8h*)(p);                  // K = koff + 0..7
  f.p.hi = *(const v8h*)(p + 8);              // K = koff + 8..15
  return f.v;
}

// B fragment (32x16 f16) from global weights W[N][K] row-major (B[k][n] = W[n][k]).
// K order used here (contiguous, same as A): lane&15 = column; lanes 0-15 hold K = k0+0..15, lanes 16-31 hold K = k0+16..31
// -> 16 contiguous halfs per lane (two b128 loads).
__device__ __forceinline__ v16h load_b_glb(const h16* __restrict__ W, int ldw,
                                           int ncol_base, int k0, int lane) {
  const int n  = ncol_base + (lane & 15);
  const int kr = k0 + ((lane >> 4) << 4);
  const h16* p = W + n * ldw + kr;
  AFrag f;
  f.p.lo = *(const v8h*)(p);
  f.p.hi = *(const v8h*)(p + 8);
  return f.v;
}

__global__ __launch_bounds__(256)
void cvt_f32_f16(const float* __restrict__ s, h16* __restrict__ d, int n) {
  int i = blockIdx.x * 256 + threadIdx.x;
  if (i < n) d[i] = (h16)s[i];
}

// C[M][N] = A[M][K](f32) x W[N][K](f16)^T + bias, fp32 out.
// Block: 256 thr = 8 waves. M-tile 16, N-tile 128 (one 16x16 WMMA tile per wave).
__global__ __launch_bounds__(256)
void proj_wmma(const float* __restrict__ A, const h16* __restrict__ Wh,
               const float* __restrict__ bias, float* __restrict__ C,
               int M, int K, int N) {
  __shared__ __attribute__((aligned(16))) h16 sh[16 * LDA];
  const int tid   = threadIdx.x;
  const int mtile = blockIdx.y * 16;
  const int ntile = blockIdx.x * 128;

  // convert one 16xK fp32 A-tile into f16 LDS; 16 threads per row
  {
    const int r = tid >> 4;
    const float* ar = A + (size_t)(mtile + r) * K;
    for (int j = tid & 15; j < K; j += 16)
      sh[r * LDA + j] = (h16)ar[j];
  }
  __syncthreads();

  const int lane  = tid & 31;
  const int w     = tid >> 5;
  const int cbase = ntile + w * 16;
  v8f acc = {};
  for (int k0 = 0; k0 < K; k0 += 32) {
    v16h a = load_a_lds(sh, k0, lane);
    v16h b = load_b_glb(Wh, K, cbase, k0, lane);
    acc = __builtin_amdgcn_wmma_f32_16x16x32_f16(false, a, false, b,
                                                 (short)0, acc, false, false);
  }
  const int mb = (lane >> 4) << 3;            // rows r / r+8 per half-wave
  const int n  = cbase + (lane & 15);
  const float bv = bias[n];
#pragma unroll
  for (int r = 0; r < 8; ++r)
    C[(size_t)(mtile + mb + r) * N + n] = acc[r] + bv;
}

// out[b,t,u,:] = tanh(encP[b,t,:] + predP[b,u,:]) @ Wout^T + bout
// Block: 64 flattened (b,t,u) rows x full V=1024 cols; 512 threads = 16 waves
// arranged 2(M) x 8(N). Each wave owns a 32x128 stripe = 2 M-subtiles x
// 8 N-subtiles: every B fragment loaded once per wave, reused for 2 WMMAs,
// and each block amortizes the 1.3 MB f16 W_out over 64 output rows
// (64 FLOP per L2 byte -> compute-bound at WMMA rates).
__global__ __launch_bounds__(512)
void joint_wmma(const float* __restrict__ encp, const float* __restrict__ predp,
                const h16* __restrict__ Wout, const float* __restrict__ bout,
                float* __restrict__ out) {
  constexpr int T = 200, U = 50, DJ = 640, V = 1024;
  __shared__ __attribute__((aligned(16))) h16 sh[64 * LDA];   // ~83 KB
  const int tid   = threadIdx.x;
  const int rbase = blockIdx.x * 64;

  // activation tile: 64 rows x 640, 8 threads per row
  {
    const int r   = tid >> 3;
    const int R   = rbase + r;
    const int b   = R / (T * U);
    const int rem = R - b * (T * U);
    const int t   = rem / U;
    const int u   = rem - t * U;
    const float* er = encp  + (size_t)(b * T + t) * DJ;
    const float* pr = predp + (size_t)(b * U + u) * DJ;
    for (int j = tid & 7; j < DJ; j += 8)
      sh[r * LDA + j] = (h16)tanhf(er[j] + pr[j]);
  }
  __syncthreads();

  const int lane = tid & 31;
  const int wave = tid >> 5;          // 0..15
  const int wm   = wave >> 3;         // 0..1  -> rows wm*32 + 0..31
  const int wn   = wave & 7;          //        -> cols wn*128 + 0..127
  const h16* shw = sh + (wm * 32) * LDA;

  v8f zero = {};
  v8f acc[2][8];
#pragma unroll
  for (int h = 0; h < 2; ++h)
#pragma unroll
    for (int s = 0; s < 8; ++s) acc[h][s] = zero;

#pragma unroll 1
  for (int k0 = 0; k0 < DJ; k0 += 32) {
    v16h a0 = load_a_lds(shw,            k0, lane);   // rows  0..15 of stripe
    v16h a1 = load_a_lds(shw + 16 * LDA, k0, lane);   // rows 16..31 of stripe
#pragma unroll
    for (int s = 0; s < 8; ++s) {
      v16h b = load_b_glb(Wout, DJ, wn * 128 + s * 16, k0, lane);
      acc[0][s] = __builtin_amdgcn_wmma_f32_16x16x32_f16(false, a0, false, b,
                                                         (short)0, acc[0][s], false, false);
      acc[1][s] = __builtin_amdgcn_wmma_f32_16x16x32_f16(false, a1, false, b,
                                                         (short)0, acc[1][s], false, false);
    }
  }

  const int mb = (lane >> 4) << 3;
#pragma unroll
  for (int h = 0; h < 2; ++h) {
#pragma unroll
    for (int s = 0; s < 8; ++s) {
      const int n = wn * 128 + s * 16 + (lane & 15);
      const float bv = bout[n];
#pragma unroll
      for (int r = 0; r < 8; ++r) {
        // streaming fp32 output (328 MB): non-temporal store, keep L2 for W_out
        __builtin_nontemporal_store(acc[h][s][r] + bv,
            out + (size_t)(rbase + wm * 32 + h * 16 + mb + r) * V + n);
      }
    }
  }
}

extern "C" void kernel_launch(void* const* d_in, const int* in_sizes, int n_in,
                              void* d_out, int out_size, void* d_ws, size_t ws_size,
                              hipStream_t stream) {
  const float* enc   = (const float*)d_in[0];   // (8,200,512)
  const float* pred  = (const float*)d_in[1];   // (8,50,640)
  const float* Wenc  = (const float*)d_in[2];   // (640,512)
  const float* benc  = (const float*)d_in[3];   // (640)
  const float* Wpred = (const float*)d_in[4];   // (640,640)
  const float* bpred = (const float*)d_in[5];   // (640)
  const float* Wout  = (const float*)d_in[6];   // (1024,640)
  const float* bout  = (const float*)d_in[7];   // (1024)
  float* out = (float*)d_out;                   // (8,200,50,1024)

  // workspace layout (bytes, 256-aligned)
  char* ws = (char*)d_ws;
  h16*   WencH  = (h16*)(ws + 0);              //  640*512*2 =  655360
  h16*   WpredH = (h16*)(ws + 655360);         //  640*640*2 =  819200
  h16*   WoutH  = (h16*)(ws + 1474560);        // 1024*640*2 = 1310720
  float* encP   = (float*)(ws + 2785280);      // 1600*640*4 = 4096000
  float* predP  = (float*)(ws + 6881280);      //  400*640*4 = 1024000

  cvt_f32_f16<<<1280, 256, 0, stream>>>(Wenc,  WencH,  640 * 512);
  cvt_f32_f16<<<1600, 256, 0, stream>>>(Wpred, WpredH, 640 * 640);
  cvt_f32_f16<<<2560, 256, 0, stream>>>(Wout,  WoutH,  1024 * 640);

  // enc projection: (1600 x 512) x (640 x 512)^T -> (1600 x 640)
  proj_wmma<<<dim3(5, 100), 256, 0, stream>>>(enc,  WencH,  benc,  encP, 1600, 512, 640);
  // pred projection: (400 x 640) x (640 x 640)^T -> (400 x 640)
  proj_wmma<<<dim3(5, 25),  256, 0, stream>>>(pred, WpredH, bpred, predP, 400, 640, 640);
  // joint: 80000 rows / 64 per block
  joint_wmma<<<1250, 512, 0, stream>>>(encP, predP, WoutH, bout, out);
}
